// KANAttention_44487271252209
// MI455X (gfx1250) — matrix operand order repaired and checked by reading, not compiled, not run on hardware
//
#include <hip/hip_runtime.h>
#include <hip/hip_bf16.h>

typedef float  v8f   __attribute__((ext_vector_type(8)));
typedef float  v2f   __attribute__((ext_vector_type(2)));
typedef __bf16 v16bf __attribute__((ext_vector_type(16)));
typedef __bf16 v8bf  __attribute__((ext_vector_type(8)));
typedef unsigned int u32x4 __attribute__((ext_vector_type(4)));
typedef int          i32x4 __attribute__((ext_vector_type(4)));
typedef int          i32x8 __attribute__((ext_vector_type(8)));

#define DIM   1024
#define SEQ   2048
#define BATCH 4
#define NUMF  16

#define KT        32                    // keys per attention iteration
#define NKT       (SEQ / KT)            // 64 iterations
#define QT_ROWS   32                    // queries per block (2 x 16-row WMMA tiles)
#define NQT       (SEQ / QT_ROWS)       // 64 query tiles
#define VROW_B    80                    // 64B of V data + 16B TDM pad (bank-conflict free)
#define VBUF_B    (DIM * VROW_B)        // 80 KB per buffer
#define SMEM_B    (2 * VBUF_B)          // double buffered: 160 KB (WGP LDS is 320 KB)

// Load 8 consecutive f32 and convert to packed bf16 (half of a 16-wide WMMA operand).
__device__ __forceinline__ v8bf load8_bf(const float* __restrict__ p) {
    const float4 f0 = *(const float4*)(p);
    const float4 f1 = *(const float4*)(p + 4);
    v8bf r;
    r[0]=(__bf16)f0.x; r[1]=(__bf16)f0.y; r[2]=(__bf16)f0.z; r[3]=(__bf16)f0.w;
    r[4]=(__bf16)f1.x; r[5]=(__bf16)f1.y; r[6]=(__bf16)f1.z; r[7]=(__bf16)f1.w;
    return r;
}

union BF16Op { v16bf v; v8bf h[2]; uint4 q[2]; };

// ---------------------------------------------------------------------------
// TDM: DMA one [1024 rows x 32 keys] bf16 V-tile (2D, row stride 2048 elems)
// from global into LDS, inserting 16B of pad every 64B so LDS rows are 80B.
// Descriptor bit layout per CDNA5 ISA.
// ---------------------------------------------------------------------------
__device__ __forceinline__ void tdm_load_vtile(unsigned long long gaddr, unsigned ldsaddr) {
    u32x4 g0;
    g0[0] = 1u;                                            // count=1, user mode
    g0[1] = ldsaddr;                                       // lds_addr [63:32]
    g0[2] = (unsigned)(gaddr & 0xffffffffu);               // global_addr[31:0]
    g0[3] = (unsigned)((gaddr >> 32) & 0x01ffffffu)        // global_addr[56:32]
          | 0x80000000u;                                   // type=2 ("image") [127:126]
    i32x8 g1;
    g1[0] = (int)(0x00010000u                              // data_size=1 (2 bytes)
          | (1u << 20)                                     // pad_enable
          | (3u << 22)                                     // pad_interval: 16 DWORDs (64B)
          | (3u << 25));                                   // pad_amount: 4 DWORDs (16B)
    g1[1] = (int)((unsigned)SEQ << 16);                    // tensor_dim0 = 2048 (bits 79:48)
    g1[2] = (int)((unsigned)DIM << 16);                    // dim0 hi=0; tensor_dim1 = 1024 (111:80)
    g1[3] = (int)((unsigned)KT << 16);                     // dim1 hi=0; tile_dim0 = 32 (127:112)
    g1[4] = DIM;                                           // tile_dim1 = 1024; tile_dim2 = 0
    g1[5] = SEQ;                                           // tensor_dim0_stride = 2048 (low 32)
    g1[6] = 0;                                             // stride hi; tensor_dim1_stride = 0
    g1[7] = 0;
    i32x4 z4 = {0, 0, 0, 0};
#if defined(__clang_major__) && (__clang_major__ >= 23)
    i32x8 z8 = {0, 0, 0, 0, 0, 0, 0, 0};
    __builtin_amdgcn_tensor_load_to_lds(g0, g1, z4, z4, z8, 0);
#else
    __builtin_amdgcn_tensor_load_to_lds(g0, g1, z4, z4, 0);
#endif
}

// ---------------------------------------------------------------------------
// K0: fold basis into Wq/Wk:  B[f][d] = sum_e basis[f][e] * W[e][d]
// ---------------------------------------------------------------------------
__global__ void fold_kernel(const float* __restrict__ basis,
                            const float* __restrict__ Wq,
                            const float* __restrict__ Wk,
                            float* __restrict__ Bq, float* __restrict__ Bk) {
    int idx   = blockIdx.x * 256 + threadIdx.x;      // 0 .. 32767
    int which = idx >> 14;
    int r     = idx & 16383;
    int f     = r >> 10;
    int d     = r & 1023;
    const float* W = which ? Wk : Wq;
    float acc = 0.f;
    for (int e = 0; e < DIM; ++e)
        acc += basis[f*DIM + e] * W[e*DIM + d];
    (which ? Bk : Bq)[f*DIM + d] = acc;
}

__global__ void foldbias_kernel(const float* __restrict__ basis,
                                const float* __restrict__ bq,
                                const float* __restrict__ bk,
                                float* __restrict__ cq, float* __restrict__ ck) {
    int idx = threadIdx.x;           // 32 threads
    int which = idx >> 4;
    int f = idx & 15;
    const float* bb = which ? bk : bq;
    float acc = 0.f;
    for (int e = 0; e < DIM; ++e) acc += basis[f*DIM + e] * bb[e];
    (which ? ck : cq)[f] = acc;
}

// ---------------------------------------------------------------------------
// K1: kan_{q,k}[b,l,f] = x[b,l,:] . B[f,:] + c[f]
// ---------------------------------------------------------------------------
__global__ void kan_kernel(const float* __restrict__ x,
                           const float* __restrict__ Bq, const float* __restrict__ Bk,
                           const float* __restrict__ cq, const float* __restrict__ ck,
                           float* __restrict__ kanq, float* __restrict__ kank) {
    int idx   = blockIdx.x * 256 + threadIdx.x;
    int which = idx >> 17;
    int r     = idx & 131071;
    int f     = r & 15;
    int l     = (r >> 4) & 2047;
    int b     = r >> 15;
    const float* B  = which ? Bk : Bq;
    const float* xr = x + ((size_t)(b*SEQ + l)) * DIM;
    const float* Br = B + f*DIM;
    float acc = which ? ck[f] : cq[f];
    for (int d = 0; d < DIM; ++d) acc += xr[d] * Br[d];
    (which ? kank : kanq)[(b*SEQ + l)*NUMF + f] = acc;
}

// ---------------------------------------------------------------------------
// K2: Vt[b][e][l] = bf16( sum_d Wv[e][d]*x[b][l][d] + bv[e] )   (M=e, N=l)
// ---------------------------------------------------------------------------
__global__ void __launch_bounds__(256) vproj_kernel(const float* __restrict__ x,
                                                    const float* __restrict__ Wv,
                                                    const float* __restrict__ bv,
                                                    __bf16* __restrict__ Vt) {
    int wid  = (blockIdx.x * 256 + threadIdx.x) >> 5;
    int lane = threadIdx.x & 31;
    int b    = wid >> 13;
    int r    = wid & 8191;
    int et   = r >> 7;
    int lt   = r & 127;
    int half = (lane >> 4) & 1;
    int l16  = lane & 15;

    int e_row = et*16 + l16;
    int l_col = lt*16 + l16;
    const float* aptr = Wv + (size_t)e_row * DIM;
    const float* bptr = x + ((size_t)(b*SEQ + l_col)) * DIM;

    v8f acc = {};
    for (int k0 = 0; k0 < DIM; k0 += 32) {
        BF16Op A;
        A.h[0] = load8_bf(aptr + k0 + half*8);
        A.h[1] = load8_bf(aptr + k0 + 16 + half*8);
        BF16Op B;
        B.h[0] = load8_bf(bptr + k0 + half*8);
        B.h[1] = load8_bf(bptr + k0 + 16 + half*8);
        acc = __builtin_amdgcn_wmma_f32_16x16x32_bf16(false, A.v, false, B.v,
                                                      (short)0, acc, false, false);
    }
    #pragma unroll
    for (int j = 0; j < 8; ++j) {
        int e = et*16 + j + half*8;
        float o = acc[j] + bv[e];
        Vt[((size_t)(b*DIM + e)) * SEQ + l_col] = (__bf16)o;
    }
}

// ---------------------------------------------------------------------------
// K3: flash attention, 32 queries/block, 32 keys/iter, TDM double-buffered V.
//  wave0: S + online softmax for query rows 0..15.
//  wave1: TDM prefetch of next V tile + S + softmax for rows 16..31.
//  all 8 waves: two P(16x32).V(32x16) updates per output tile, sharing the
//               V B-operand registers; full-K v_wmma_f32_16x16x32_bf16.
// ---------------------------------------------------------------------------
__global__ void __launch_bounds__(256) attn_kernel(const float* __restrict__ kanq,
                                                   const float* __restrict__ kank,
                                                   const __bf16* __restrict__ Vt,
                                                   float* __restrict__ out) {
    extern __shared__ __align__(16) char smem[];          // 2 x 80KB V tile buffers
    __shared__ __align__(16) __bf16 lds_p[QT_ROWS*KT];    // P: 32 rows x 32 keys
    __shared__ float lds_corr[QT_ROWS];
    __shared__ float lds_linv[QT_ROWS];

    const int b    = blockIdx.x >> 6;                     // 4 batches
    const int qt   = blockIdx.x & 63;                     // 64 query tiles of 32
    const int tid  = threadIdx.x;
    const int lane = tid & 31;
    const int wv   = tid >> 5;
    const int half = (lane >> 4) & 1;
    const int l16  = lane & 15;
    const float scale = 0.03125f;   // 1/sqrt(1024)

    const unsigned lds_base = (unsigned)(unsigned long long)(uintptr_t)smem;
    const unsigned long long vt_base =
        (unsigned long long)(uintptr_t)(Vt + (size_t)b * DIM * SEQ);

    const v8f zero8 = {};
    v8f acc0[8], acc1[8];                                 // 2 query tiles x 8 d tiles
    #pragma unroll
    for (int t = 0; t < 8; ++t) { acc0[t] = zero8; acc1[t] = zero8; }

    // waves 0/1 each own the softmax state for one 16-row query sub-tile
    v2f  aq[4];
    float m[8], lsum[8];
    if (wv <= 1) {
        const float* kq = kanq + ((size_t)(b*SEQ + qt*QT_ROWS + wv*16 + l16)) * NUMF + half*2;
        #pragma unroll
        for (int kc = 0; kc < 4; ++kc) aq[kc] = *(const v2f*)(kq + kc*4);
        #pragma unroll
        for (int j = 0; j < 8; ++j) { m[j] = -1e30f; lsum[j] = 0.f; }
    }
    if (wv == 1) {                                        // prefetch tile 0
        tdm_load_vtile(vt_base, lds_base);
    }

    for (int kt = 0; kt < NKT; ++kt) {
        if (wv <= 1) {
            if (wv == 1 && kt + 1 < NKT) {                // prefetch next tile first
                tdm_load_vtile(vt_base + (unsigned long long)(kt + 1) * (KT * 2),
                               lds_base + ((kt + 1) & 1) * VBUF_B);
            }
            // S(16x32) for this wave's query sub-tile via 8x f32 WMMA (K=16)
            v8f s0 = zero8, s1 = zero8;
            const float* kk = kank + ((size_t)(b*SEQ + kt*KT + l16)) * NUMF + half*2;
            #pragma unroll
            for (int kc = 0; kc < 4; ++kc) {
                v2f bb = *(const v2f*)(kk + kc*4);
                s0 = __builtin_amdgcn_wmma_f32_16x16x4_f32(false, aq[kc], false, bb,
                                                           (short)0, s0, false, false);
            }
            const float* kk1 = kk + 16*NUMF;
            #pragma unroll
            for (int kc = 0; kc < 4; ++kc) {
                v2f bb = *(const v2f*)(kk1 + kc*4);
                s1 = __builtin_amdgcn_wmma_f32_16x16x4_f32(false, aq[kc], false, bb,
                                                           (short)0, s1, false, false);
            }
            #pragma unroll
            for (int j = 0; j < 8; ++j) {
                float sv0 = s0[j] * scale;
                float sv1 = s1[j] * scale;
                float rmax = fmaxf(sv0, sv1);
                rmax = fmaxf(rmax, __shfl_xor(rmax, 1, 16));
                rmax = fmaxf(rmax, __shfl_xor(rmax, 2, 16));
                rmax = fmaxf(rmax, __shfl_xor(rmax, 4, 16));
                rmax = fmaxf(rmax, __shfl_xor(rmax, 8, 16));
                float mn   = fmaxf(m[j], rmax);
                float corr = __expf(m[j] - mn);
                float p0   = __expf(sv0 - mn);
                float p1   = __expf(sv1 - mn);
                float rs = p0 + p1;
                rs += __shfl_xor(rs, 1, 16);
                rs += __shfl_xor(rs, 2, 16);
                rs += __shfl_xor(rs, 4, 16);
                rs += __shfl_xor(rs, 8, 16);
                lsum[j] = lsum[j]*corr + rs;
                m[j]    = mn;
                int row = wv*16 + j + half*8;             // 0..31
                lds_p[row*KT + l16]      = (__bf16)p0;    // keys kt*32 + 0..15
                lds_p[row*KT + 16 + l16] = (__bf16)p1;    // keys kt*32 + 16..31
                if (l16 == 0) lds_corr[row] = corr;
            }
            if (wv == 1) {
                if (kt + 1 < NKT) __builtin_amdgcn_s_wait_tensorcnt(1);
                else              __builtin_amdgcn_s_wait_tensorcnt(0);
            }
        }
        __syncthreads();

        // P A-operands for both query sub-tiles (full 16x32, no padding)
        BF16Op pa0, pa1;
        pa0.q[0] = *(const uint4*)&lds_p[l16*KT + half*8];
        pa0.q[1] = *(const uint4*)&lds_p[l16*KT + 16 + half*8];
        pa1.q[0] = *(const uint4*)&lds_p[(16 + l16)*KT + half*8];
        pa1.q[1] = *(const uint4*)&lds_p[(16 + l16)*KT + 16 + half*8];
        float cj0[8], cj1[8];
        #pragma unroll
        for (int j = 0; j < 8; ++j) {
            cj0[j] = lds_corr[j + half*8];
            cj1[j] = lds_corr[16 + j + half*8];
        }

        const char* vbuf = smem + (kt & 1) * VBUF_B;
        #pragma unroll
        for (int t = 0; t < 8; ++t) {
            int d = wv*128 + t*16 + l16;                  // output column (B operand col)
            const char* rp = vbuf + (size_t)d * VROW_B + half*16;
            BF16Op pb;                                    // 32x16 B: lanes<16 K0..7,16..23; >=16 K8..15,24..31
            pb.q[0] = *(const uint4*)rp;
            pb.q[1] = *(const uint4*)(rp + 32);
            #pragma unroll
            for (int j = 0; j < 8; ++j) { acc0[t][j] *= cj0[j]; acc1[t][j] *= cj1[j]; }
            acc0[t] = __builtin_amdgcn_wmma_f32_16x16x32_bf16(false, pa0.v, false, pb.v,
                                                              (short)0, acc0[t], false, false);
            acc1[t] = __builtin_amdgcn_wmma_f32_16x16x32_bf16(false, pa1.v, false, pb.v,
                                                              (short)0, acc1[t], false, false);
        }
        __syncthreads();
    }

    if (wv <= 1 && l16 == 0) {
        #pragma unroll
        for (int j = 0; j < 8; ++j) lds_linv[wv*16 + j + half*8] = 1.0f / lsum[j];
    }
    __syncthreads();

    float inv0[8], inv1[8];
    #pragma unroll
    for (int j = 0; j < 8; ++j) {
        inv0[j] = lds_linv[j + half*8];
        inv1[j] = lds_linv[16 + j + half*8];
    }
    #pragma unroll
    for (int t = 0; t < 8; ++t) {
        #pragma unroll
        for (int j = 0; j < 8; ++j) {
            int row0 = qt*QT_ROWS + j + half*8;
            int row1 = row0 + 16;
            int col  = wv*128 + t*16 + l16;
            out[((size_t)(b*SEQ + row0)) * DIM + col] = acc0[t][j] * inv0[j];
            out[((size_t)(b*SEQ + row1)) * DIM + col] = acc1[t][j] * inv1[j];
        }
    }
}

// ---------------------------------------------------------------------------
extern "C" void kernel_launch(void* const* d_in, const int* in_sizes, int n_in,
                              void* d_out, int out_size, void* d_ws, size_t ws_size,
                              hipStream_t stream) {
    const float* x     = (const float*)d_in[0];
    const float* basis = (const float*)d_in[1];
    const float* Wq    = (const float*)d_in[2];
    const float* bq    = (const float*)d_in[3];
    const float* Wk    = (const float*)d_in[4];
    const float* bk    = (const float*)d_in[5];
    const float* Wv    = (const float*)d_in[6];
    const float* bv    = (const float*)d_in[7];
    float* out = (float*)d_out;

    char* ws = (char*)d_ws;
    size_t off = 0;
    __bf16* Vt = (__bf16*)(ws + off); off += (size_t)BATCH*DIM*SEQ*2;   // 16 MB
    float* Bq  = (float*)(ws + off);  off += (size_t)NUMF*DIM*4;
    float* Bk  = (float*)(ws + off);  off += (size_t)NUMF*DIM*4;
    float* cq  = (float*)(ws + off);  off += 64;
    float* ck  = (float*)(ws + off);  off += 64;
    float* kanq = (float*)(ws + off); off += (size_t)BATCH*SEQ*NUMF*4;
    float* kank = (float*)(ws + off); off += (size_t)BATCH*SEQ*NUMF*4;

    // Allow 160KB dynamic LDS (CDNA5 WGP has 320KB).
    (void)hipFuncSetAttribute((const void*)attn_kernel,
                              hipFuncAttributeMaxDynamicSharedMemorySize, SMEM_B);

    fold_kernel<<<128, 256, 0, stream>>>(basis, Wq, Wk, Bq, Bk);
    foldbias_kernel<<<1, 32, 0, stream>>>(basis, bq, bk, cq, ck);
    kan_kernel<<<1024, 256, 0, stream>>>(x, Bq, Bk, cq, ck, kanq, kank);
    vproj_kernel<<<4096, 256, 0, stream>>>(x, Wv, bv, Vt);
    attn_kernel<<<BATCH*NQT, 256, SMEM_B, stream>>>(kanq, kank, Vt, out);
}
